// S4DClassifier1D_4853313044575
// MI455X (gfx1250) — matrix operand order repaired and checked by reading, not compiled, not run on hardware
//
#include <hip/hip_runtime.h>
#include <hip/hip_bf16.h>
#include <math.h>

// ---------------- problem constants ----------------
#define BB    16
#define TT    4096
#define CC    144
#define NN    64
#define DEPTH 6
#define HID   576
#define NCLS  5
#define EPSV  1e-5f

typedef __attribute__((ext_vector_type(16))) _Float16 v16h;
typedef __attribute__((ext_vector_type(8)))  float    v8f;

// fragment sizes (halves)
#define F1_PER_LAYER (36*5*32*16)   // fc1 B-frags: 36 N-tiles x 5 K-chunks
#define F2_PER_LAYER (9*18*32*16)   // fc2 B-frags: 9 N-tiles x 18 K-chunks

__device__ __forceinline__ float gelu_exact(float x) {
    return 0.5f * x * (1.0f + erff(x * 0.70710678118654752440f));
}
__device__ __forceinline__ float softplusf(float x) {
    return (x > 20.0f) ? x : log1pf(expf(x));
}

// ---------------- K1: conv1d(k=5,pad=2) + GELU -> h (B,T,C) ----------------
__global__ void embed_kernel(const float* __restrict__ x,
                             const float* __restrict__ w,
                             float* __restrict__ h) {
    size_t id = (size_t)blockIdx.x * blockDim.x + threadIdx.x;
    if (id >= (size_t)BB * TT * CC) return;
    int c = (int)(id % CC);
    int t = (int)((id / CC) % TT);
    int b = (int)(id / ((size_t)CC * TT));
    float s = 0.0f;
#pragma unroll
    for (int k = 0; k < 5; ++k) {
        int tt = t + k - 2;
        if (tt >= 0 && tt < TT) s = fmaf(w[c * 5 + k], x[(size_t)b * TT + tt], s);
    }
    h[id] = gelu_exact(s);
}

// ---------------- LN over C per (b,t); block = 256, grid = B*T ----------------
// writes transposed u_t (B,C,T) for coalesced scan
__global__ void ln1_kernel(const float* __restrict__ h,
                           const float* __restrict__ g,
                           const float* __restrict__ bta,
                           float* __restrict__ u_t) {
    int bt = blockIdx.x;
    int b  = bt / TT, t = bt % TT;
    int c  = threadIdx.x;
    float v = (c < CC) ? h[(size_t)bt * CC + c] : 0.0f;
    __shared__ float s1[256], s2[256];
    s1[c] = v; s2[c] = v * v;
    __syncthreads();
    for (int off = 128; off > 0; off >>= 1) {
        if (c < off) { s1[c] += s1[c + off]; s2[c] += s2[c + off]; }
        __syncthreads();
    }
    float mean = s1[0] * (1.0f / CC);
    float var  = s2[0] * (1.0f / CC) - mean * mean;
    float inv  = rsqrtf(var + EPSV);
    if (c < CC) {
        float hn = (v - mean) * inv * g[c] + bta[c];
        u_t[((size_t)(b * CC + c)) * TT + t] = hn;
    }
}

// ---------------- K3: diagonal SSM scan; one wave per (b,c) ----------------
// y[t] = sum_n mix[c,n]*S_n[t] + Dskip[c]*u[t];  S_n = r_n*S_n + u
__global__ void ssm_scan_kernel(const float* __restrict__ u_t,
                                float* __restrict__ y_t,
                                const float* __restrict__ mix,
                                const float* __restrict__ log_tau,
                                const float* __restrict__ log_dt,
                                const float* __restrict__ Dskip,
                                int layer) {
    int wid  = (blockIdx.x * blockDim.x + threadIdx.x) >> 5;
    int lane = threadIdx.x & 31;
    if (wid >= BB * CC) return;
    int b = wid / CC, c = wid % CC;

    float dt   = softplusf(log_dt[layer]);
    float tau1 = softplusf(log_tau[layer * NN + lane]);
    float tau2 = softplusf(log_tau[layer * NN + lane + 32]);
    float r1 = expf(-tau1 * dt);
    float r2 = expf(-tau2 * dt);
    float m1 = mix[((size_t)layer * CC + c) * NN + lane];
    float m2 = mix[((size_t)layer * CC + c) * NN + lane + 32];
    float dsk = Dskip[layer * CC + c];

    const float* up = u_t + ((size_t)(b * CC + c)) * TT;
    float*       yp = y_t + ((size_t)(b * CC + c)) * TT;

    float s1 = 0.0f, s2 = 0.0f;
    for (int tb = 0; tb < TT / 32; ++tb) {
        float uc = up[tb * 32 + lane];   // coalesced
        float yv = 0.0f;
#pragma unroll 4
        for (int s = 0; s < 32; ++s) {
            float uu = __shfl(uc, s);
            s1 = fmaf(r1, s1, uu);
            s2 = fmaf(r2, s2, uu);
            float p = fmaf(m1, s1, m2 * s2);
#pragma unroll
            for (int off = 16; off > 0; off >>= 1) p += __shfl_xor(p, off);
            if (lane == s) yv = p + dsk * uu;
        }
        yp[tb * 32 + lane] = yv;         // coalesced
    }
}

// ---------------- K4: h += y^T ; LN2 ; emit f16 activations ----------------
__global__ void addln2_kernel(float* __restrict__ h,
                              const float* __restrict__ y_t,
                              const float* __restrict__ g,
                              const float* __restrict__ bta,
                              _Float16* __restrict__ hn2) {
    int bt = blockIdx.x;
    int b  = bt / TT, t = bt % TT;
    int c  = threadIdx.x;
    float v = 0.0f;
    if (c < CC) {
        v = h[(size_t)bt * CC + c] + y_t[((size_t)(b * CC + c)) * TT + t];
        h[(size_t)bt * CC + c] = v;   // residual update in place
    }
    __shared__ float s1[256], s2[256];
    s1[c] = v; s2[c] = v * v;
    __syncthreads();
    for (int off = 128; off > 0; off >>= 1) {
        if (c < off) { s1[c] += s1[c + off]; s2[c] += s2[c + off]; }
        __syncthreads();
    }
    float mean = s1[0] * (1.0f / CC);
    float var  = s2[0] * (1.0f / CC) - mean * mean;
    float inv  = rsqrtf(var + EPSV);
    if (c < CC) {
        float hn = (v - mean) * inv * g[c] + bta[c];
        hn2[(size_t)bt * CC + c] = (_Float16)hn;
    }
}

// ---------------- weight packing into WMMA B-fragment layout ----------------
// B(32x16 f16): lanes 0-15 hold K {0..7,16..23}, lanes 16-31 K {8..15,24..31}
__global__ void pack_w1_kernel(const float* __restrict__ fc1_w,
                               _Float16* __restrict__ w1f) {
    size_t idx = (size_t)blockIdx.x * blockDim.x + threadIdx.x;
    if (idx >= (size_t)DEPTH * F1_PER_LAYER) return;
    int layer = (int)(idx / F1_PER_LAYER);
    int r     = (int)(idx % F1_PER_LAYER);
    int ntile = r / (5 * 512);
    int r2    = r % (5 * 512);
    int kc    = r2 / 512;
    int lane  = (r2 % 512) / 16;
    int hh    = r2 % 16;
    int K = kc * 32 + ((lane < 16) ? 0 : 8) + (hh & 7) + ((hh & 8) ? 16 : 0);
    int Nn = ntile * 16 + (lane & 15);
    float v = (K < CC) ? fc1_w[((size_t)layer * HID + Nn) * CC + K] : 0.0f;
    w1f[idx] = (_Float16)v;
}

__global__ void pack_w2_kernel(const float* __restrict__ fc2_w,
                               _Float16* __restrict__ w2f) {
    size_t idx = (size_t)blockIdx.x * blockDim.x + threadIdx.x;
    if (idx >= (size_t)DEPTH * F2_PER_LAYER) return;
    int layer = (int)(idx / F2_PER_LAYER);
    int r     = (int)(idx % F2_PER_LAYER);
    int ntile = r / (18 * 512);
    int r2    = r % (18 * 512);
    int kc    = r2 / 512;
    int lane  = (r2 % 512) / 16;
    int hh    = r2 % 16;
    int K = kc * 32 + ((lane < 16) ? 0 : 8) + (hh & 7) + ((hh & 8) ? 16 : 0);     // K < 576 always
    int Nn = ntile * 16 + (lane & 15);                 // Nn < 144 always
    w2f[idx] = (_Float16)fc2_w[((size_t)layer * CC + Nn) * HID + K];
}

// ---------------- K5: fused MLP  h += gelu(hn2@W1^T+b1)@W2^T+b2 ----------------
// block: 256 threads (8 waves), tile M=32 rows; intermediate lives in LDS.
__global__ void mlp_fused_kernel(const _Float16* __restrict__ hn2,
                                 const _Float16* __restrict__ w1f,
                                 const _Float16* __restrict__ w2f,
                                 const float* __restrict__ b1,
                                 const float* __restrict__ b2,
                                 float* __restrict__ h) {
    __shared__ alignas(32) _Float16 Afrag[2 * 5  * 32 * 16];  // 10 KB: A tile, K padded 144->160
    __shared__ alignas(32) _Float16 Mfrag[2 * 18 * 32 * 16];  // 36 KB: gelu(mid) as A-frags for GEMM2

    const int tid  = threadIdx.x;
    const int wave = tid >> 5;
    const int lane = tid & 31;
    const int row0 = blockIdx.x * 32;

    // stage A tile into ISA A-matrix fragment layout:
    // lane<16: K {0..7,16..23}; lane>=16: K {8..15,24..31}; M = msub*16 + lane%16
    for (int idx = tid; idx < 2 * 5 * 512; idx += 256) {
        int msub = idx / (5 * 512);
        int r    = idx % (5 * 512);
        int kc   = r / 512;
        int ln   = (r % 512) / 16;
        int hh   = r % 16;
        int K = kc * 32 + ((hh < 8) ? (hh + ((ln < 16) ? 0 : 8))
                                    : ((hh - 8) + 16 + ((ln < 16) ? 0 : 8)));
        int M = msub * 16 + (ln & 15);
        Afrag[idx] = (K < CC) ? hn2[(size_t)(row0 + M) * CC + K] : (_Float16)0.0f;
    }
    __syncthreads();

    // GEMM1: (32 x 160) x (160 x 576) ; 2 x 36 = 72 output tiles over 8 waves
    for (int j = wave; j < 72; j += 8) {
        int msub = j / 36, ntile = j % 36;
        float bias = b1[ntile * 16 + (lane & 15)];
        v8f acc;
#pragma unroll
        for (int e = 0; e < 8; ++e) acc[e] = bias;
#pragma unroll
        for (int kc = 0; kc < 5; ++kc) {
            v16h a = *(const v16h*)&Afrag[((msub * 5 + kc) * 32 + lane) * 16];
            v16h bm = *(const v16h*)&w1f[(((size_t)ntile * 5 + kc) * 32 + lane) * 16];
            acc = __builtin_amdgcn_wmma_f32_16x16x32_f16(false, a, false, bm,
                                                         (short)0, acc, false, false);
        }
        // GELU, convert to f16, scatter into A-fragment layout for GEMM2
#pragma unroll
        for (int rr = 0; rr < 8; ++rr) {
            float v = gelu_exact(acc[rr]);
            int M  = msub * 16 + rr + ((lane < 16) ? 0 : 8);
            int K2 = ntile * 16 + (lane & 15);
            int kc2 = K2 >> 5;
            int kk  = K2 & 31;
            int ln2 = (M & 15) + ((kk & 8) ? 16 : 0);
            int hh2 = ((kk & 16) ? 8 : 0) + (kk & 7);
            Mfrag[(((M >> 4) * 18 + kc2) * 32 + ln2) * 16 + hh2] = (_Float16)v;
        }
    }
    __syncthreads();

    // GEMM2: (32 x 576) x (576 x 144) ; 2 x 9 = 18 output tiles over 8 waves
    for (int j = wave; j < 18; j += 8) {
        int msub = j / 9, ntile = j % 9;
        float bias = b2[ntile * 16 + (lane & 15)];
        v8f acc;
#pragma unroll
        for (int e = 0; e < 8; ++e) acc[e] = bias;
#pragma unroll
        for (int kc = 0; kc < 18; ++kc) {
            v16h a = *(const v16h*)&Mfrag[((msub * 18 + kc) * 32 + lane) * 16];
            v16h bm = *(const v16h*)&w2f[(((size_t)ntile * 18 + kc) * 32 + lane) * 16];
            acc = __builtin_amdgcn_wmma_f32_16x16x32_f16(false, a, false, bm,
                                                         (short)0, acc, false, false);
        }
#pragma unroll
        for (int rr = 0; rr < 8; ++rr) {
            int M = msub * 16 + rr + ((lane < 16) ? 0 : 8);
            int n = ntile * 16 + (lane & 15);
            size_t o = (size_t)(row0 + M) * CC + n;
            h[o] += acc[rr];          // residual add (unique writer per element)
        }
    }
}

// ---------------- K6: final LN + mean pool (atomics) ----------------
__global__ void finalln_pool_kernel(const float* __restrict__ h,
                                    const float* __restrict__ g,
                                    const float* __restrict__ bta,
                                    float* __restrict__ pooled) {
    int bt = blockIdx.x;
    int b  = bt / TT;
    int c  = threadIdx.x;
    float v = (c < CC) ? h[(size_t)bt * CC + c] : 0.0f;
    __shared__ float s1[256], s2[256];
    s1[c] = v; s2[c] = v * v;
    __syncthreads();
    for (int off = 128; off > 0; off >>= 1) {
        if (c < off) { s1[c] += s1[c + off]; s2[c] += s2[c + off]; }
        __syncthreads();
    }
    float mean = s1[0] * (1.0f / CC);
    float var  = s2[0] * (1.0f / CC) - mean * mean;
    float inv  = rsqrtf(var + EPSV);
    if (c < CC) {
        float hn = (v - mean) * inv * g[c] + bta[c];
        atomicAdd(&pooled[b * CC + c], hn * (1.0f / TT));
    }
}

// ---------------- K7: head ----------------
__global__ void head_kernel(const float* __restrict__ pooled,
                            const float* __restrict__ hw,
                            const float* __restrict__ hb,
                            float* __restrict__ out) {
    int idx = threadIdx.x;
    if (idx >= BB * NCLS) return;
    int b = idx / NCLS, j = idx % NCLS;
    float s = hb[j];
    for (int c = 0; c < CC; ++c) s = fmaf(pooled[b * CC + c], hw[j * CC + c], s);
    out[idx] = s;
}

// ---------------- host launch ----------------
extern "C" void kernel_launch(void* const* d_in, const int* in_sizes, int n_in,
                              void* d_out, int out_size, void* d_ws, size_t ws_size,
                              hipStream_t stream) {
    const float* x       = (const float*)d_in[0];
    const float* embed_w = (const float*)d_in[1];
    const float* ln1_g   = (const float*)d_in[2];
    const float* ln1_b   = (const float*)d_in[3];
    const float* mix     = (const float*)d_in[4];
    const float* log_tau = (const float*)d_in[5];
    const float* log_dt  = (const float*)d_in[6];
    const float* Dskip   = (const float*)d_in[7];
    const float* ln2_g   = (const float*)d_in[8];
    const float* ln2_b   = (const float*)d_in[9];
    const float* fc1_w   = (const float*)d_in[10];
    const float* fc1_b   = (const float*)d_in[11];
    const float* fc2_w   = (const float*)d_in[12];
    const float* fc2_b   = (const float*)d_in[13];
    const float* normf_g = (const float*)d_in[14];
    const float* normf_b = (const float*)d_in[15];
    const float* head_w  = (const float*)d_in[16];
    const float* head_b  = (const float*)d_in[17];
    float* out = (float*)d_out;

    // workspace carve-up (all 256B aligned)
    char* ws = (char*)d_ws;
    const size_t szHT = (size_t)BB * TT * CC * sizeof(float);   // 37,748,736
    float*    h    = (float*)ws;                 ws += szHT;
    float*    u_t  = (float*)ws;                 /* aliased below */ 
    _Float16* hn2  = (_Float16*)ws;              ws += szHT;    // u_t reused as hn2 after scan
    float*    y_t  = (float*)ws;                 ws += szHT;
    _Float16* w1f  = (_Float16*)ws;              ws += (size_t)DEPTH * F1_PER_LAYER * sizeof(_Float16);
    _Float16* w2f  = (_Float16*)ws;              ws += (size_t)DEPTH * F2_PER_LAYER * sizeof(_Float16);
    float*    pooled = (float*)ws;               ws += (size_t)BB * CC * sizeof(float);

    // pack weights into WMMA fragment layout (once per call)
    {
        int n1 = DEPTH * F1_PER_LAYER;
        pack_w1_kernel<<<(n1 + 255) / 256, 256, 0, stream>>>(fc1_w, w1f);
        int n2 = DEPTH * F2_PER_LAYER;
        pack_w2_kernel<<<(n2 + 255) / 256, 256, 0, stream>>>(fc2_w, w2f);
    }

    // embed conv + gelu
    {
        size_t n = (size_t)BB * TT * CC;
        embed_kernel<<<(unsigned)((n + 255) / 256), 256, 0, stream>>>(x, embed_w, h);
    }

    const int nBT = BB * TT;  // 65536
    for (int i = 0; i < DEPTH; ++i) {
        ln1_kernel<<<nBT, 256, 0, stream>>>(h, ln1_g + i * CC, ln1_b + i * CC, u_t);
        ssm_scan_kernel<<<(BB * CC) / 8, 256, 0, stream>>>(u_t, y_t, mix, log_tau,
                                                           log_dt, Dskip, i);
        addln2_kernel<<<nBT, 256, 0, stream>>>(h, y_t, ln2_g + i * CC, ln2_b + i * CC, hn2);
        mlp_fused_kernel<<<nBT / 32, 256, 0, stream>>>(hn2,
                                                       w1f + (size_t)i * F1_PER_LAYER,
                                                       w2f + (size_t)i * F2_PER_LAYER,
                                                       fc1_b + i * HID,
                                                       fc2_b + i * CC, h);
    }

    hipMemsetAsync(pooled, 0, (size_t)BB * CC * sizeof(float), stream);
    finalln_pool_kernel<<<nBT, 256, 0, stream>>>(h, normf_g, normf_b, pooled);
    head_kernel<<<1, 128, 0, stream>>>(pooled, head_w, head_b, out);
}
